// GraphActor_17300128268642
// MI455X (gfx1250) — matrix operand order repaired and checked by reading, not compiled, not run on hardware
//
#include <hip/hip_runtime.h>

typedef __attribute__((ext_vector_type(16))) __bf16 v16bf;
typedef __attribute__((ext_vector_type(8)))  float  v8f;

#define N_NODES 13
#define TILE 16

// Packed bf16 weight offsets in d_ws (elements). Order:
// 0 base_w1(15,128) 1 base_w2(128,64) 2 hip_w1 3 hip_w2 4 thigh_w1 5 thigh_w2
// 6 calf_w1 7 calf_w2 8..11 gn1 wk,wq,wv,ws(128,128) 12..15 gn2 wk,wq,wv,ws(128,64)
// 16 leg_w1(256,256) 17 leg_w2(256,128) 18 leg_w3(128,3->16)
__constant__ int WOFF[19] = {0,4096,12288,16384,24576,28672,36864,40960,
                             49152,65536,81920,98304,114688,122880,131072,139264,
                             147456,212992,245760};

__constant__ int NBRS[13][4] = {
  {1,2,3,4},{0,5,0,0},{0,6,0,0},{0,7,0,0},{0,8,0,0},
  {1,9,0,0},{2,10,0,0},{3,11,0,0},{4,12,0,0},
  {5,0,0,0},{6,0,0,0},{7,0,0,0},{8,0,0,0}};
__constant__ int NBRC[13] = {4,2,2,2,2,2,2,2,2,1,1,1,1};

struct KParams {
  const float* obs;
  const float* latent;
  const __bf16* wpk;
  const float* bias[19];
  float* out;
};

__device__ __forceinline__ v8f zero8() {
  v8f c = {0.f,0.f,0.f,0.f,0.f,0.f,0.f,0.f};
  return c;
}

__device__ __forceinline__ v8f bwmma(v16bf a, v16bf b, v8f c) {
  return __builtin_amdgcn_wmma_f32_16x16x32_bf16(false, a, false, b, (short)0, c,
                                                 false, false);
}

// A fragment from row-major bf16 [16 x lda] tile: lane = hi*16+m holds
// K = kt*32 + hi*8 + [0..7]  (elems 0..7)  and K = kt*32 + 16 + hi*8 + [0..7]
__device__ __forceinline__ v16bf load_a(const __bf16* Abase, int lda, int m, int kt, int hi) {
  const __bf16* p0 = Abase + m * lda + kt * 32 + hi * 8;
  v16bf a;
  #pragma unroll
  for (int i = 0; i < 8; ++i) { a[i] = p0[i]; a[i + 8] = p0[16 + i]; }
  return a;
}

// Packed B fragment: one contiguous 32B load per lane.
__device__ __forceinline__ v16bf load_w(const __bf16* wpk, int widx, int NT,
                                        int kt, int nt, int lane) {
  return *(const v16bf*)(wpk + WOFF[widx] + ((size_t)((kt * NT + nt) * 32 + lane) << 4));
}

__device__ __forceinline__ float eluf(float x) { return x > 0.f ? x : (__expf(x) - 1.f); }
__device__ __forceinline__ float sigm(float x) { return 1.f / (1.f + __expf(-x)); }

// ResGatedGraphConv: out = x@Ws + bs + sum_src sigmoid(k[dst]+q[src]) * v[src]
// x is bf16 LDS [13][16][128]; output (optionally ELU) written back into X[:, :, 0:O].
template<int NT, bool DO_ELU>
__device__ void conv_layer(const KParams& p, __bf16* X, __bf16* Q, __bf16* V, float* S,
                           int kw, int qw, int vw, int sw) {
  const int tid = threadIdx.x;
  const int lane = tid & 31, wid = tid >> 5;
  const int hi = lane >> 4, mn = lane & 15;
  const int O = NT * 16;
  const int NITEM = N_NODES * NT;
  // ---- C1: q, v (bf16), s = x@Ws+bs (f32)
  for (int item = wid; item < 3 * NITEM; item += 8) {
    int mat = item / NITEM;
    int r2 = item - mat * NITEM;
    int node = r2 / NT, nt = r2 - node * NT;
    int widx = (mat == 0) ? qw : (mat == 1 ? vw : sw);
    v8f c = zero8();
    const __bf16* A = X + node * TILE * 128;
    #pragma unroll
    for (int kt = 0; kt < 4; ++kt)
      c = bwmma(load_a(A, 128, mn, kt, hi), load_w(p.wpk, widx, NT, kt, nt, lane), c);
    float bb = p.bias[widx][nt * 16 + mn];
    #pragma unroll
    for (int r = 0; r < 8; ++r) {
      float val = c[r] + bb;
      int off = (node * TILE + r + 8 * hi) * O + nt * 16 + mn;
      if (mat == 0)      Q[off] = (__bf16)val;
      else if (mat == 1) V[off] = (__bf16)val;
      else               S[off] = val;
    }
  }
  __syncthreads();
  // ---- C2: k[dst] in registers + gated scatter-add into S (wave-exclusive per (d,nt))
  for (int item = wid; item < NITEM; item += 8) {
    int d = item / NT, nt = item - d * NT;
    v8f c = zero8();
    const __bf16* A = X + d * TILE * 128;
    #pragma unroll
    for (int kt = 0; kt < 4; ++kt)
      c = bwmma(load_a(A, 128, mn, kt, hi), load_w(p.wpk, kw, NT, kt, nt, lane), c);
    float kb = p.bias[kw][nt * 16 + mn];
    int col = nt * 16 + mn;
    int cnt = NBRC[d];
    #pragma unroll
    for (int r = 0; r < 8; ++r) {
      int m = r + 8 * hi;
      float kd = c[r] + kb;
      float acc = S[(d * TILE + m) * O + col];
      for (int j = 0; j < cnt; ++j) {
        int src = NBRS[d][j];
        float qv = (float)Q[(src * TILE + m) * O + col];
        float vv = (float)V[(src * TILE + m) * O + col];
        acc += sigm(kd + qv) * vv;
      }
      S[(d * TILE + m) * O + col] = acc;
    }
  }
  __syncthreads();
  // ---- C3: X = (elu?) S as bf16
  for (int idx = tid; idx < N_NODES * TILE * O; idx += 256) {
    float val = S[idx];
    if (DO_ELU) val = eluf(val);
    int nm = idx / O, cc = idx - nm * O;
    X[nm * 128 + cc] = (__bf16)val;
  }
  __syncthreads();
}

__global__ void __launch_bounds__(256)
actor_kernel(KParams p, int B) {
  __shared__ __bf16 X[N_NODES * TILE * 128];   // node features / conv io (53 KB)
  __shared__ __bf16 Q[N_NODES * TILE * 128];   // q / hidden / leg buffers (53 KB)
  __shared__ __bf16 V[N_NODES * TILE * 128];   // v / leg buffers (53 KB)
  __shared__ float  S[N_NODES * TILE * 128];   // f32 accumulator (104 KB)

  const int tid = threadIdx.x;
  const int lane = tid & 31, wid = tid >> 5;
  const int hi = lane >> 4, mn = lane & 15;
  const int b0 = blockIdx.x * TILE;
  if (b0 >= B) return;

  // ================= Encoders =================
  // E1: hidden = elu(feat @ W1 + b1) -> Q [13][16][128]
  for (int item = wid; item < N_NODES * 8; item += 8) {
    int node = item >> 3, nt = item & 7;
    int widx, fbase, K;
    if (node == 0)      { widx = 0; fbase = 0;                      K = 15; }
    else if (node < 5)  { widx = 2; fbase = 15 + 24 * (node - 1);   K = 8;  }
    else if (node < 9)  { widx = 4; fbase = 15 + 24 * (node - 5) + 8;  K = 8; }
    else                { widx = 6; fbase = 15 + 24 * (node - 9) + 16; K = 8; }
    const float* orow = p.obs + (size_t)(b0 + mn) * 111 + fbase;
    v16bf a;
    #pragma unroll
    for (int e = 0; e < 8; ++e) {
      int k0 = hi * 8 + e;
      int k1 = 16 + hi * 8 + e;
      a[e]     = (k0 < K) ? (__bf16)orow[k0] : (__bf16)0.0f;
      a[e + 8] = (k1 < K) ? (__bf16)orow[k1] : (__bf16)0.0f;
    }
    v16bf w = load_w(p.wpk, widx, 8, 0, nt, lane);
    v8f c = zero8();
    c = bwmma(a, w, c);
    float bb = p.bias[widx][nt * 16 + mn];
    #pragma unroll
    for (int r = 0; r < 8; ++r)
      Q[(node * TILE + r + 8 * hi) * 128 + nt * 16 + mn] = (__bf16)eluf(c[r] + bb);
  }
  __syncthreads();
  // E2: nodes = hidden @ W2 + b2 -> X[:, :, 0:64]; latent -> X[:, :, 64:128]
  for (int item = wid; item < N_NODES * 4; item += 8) {
    int node = item >> 2, nt = item & 3;
    int widx = (node == 0) ? 1 : (node < 5 ? 3 : (node < 9 ? 5 : 7));
    v8f c = zero8();
    const __bf16* A = Q + node * TILE * 128;
    #pragma unroll
    for (int kt = 0; kt < 4; ++kt)
      c = bwmma(load_a(A, 128, mn, kt, hi), load_w(p.wpk, widx, 4, kt, nt, lane), c);
    float bb = p.bias[widx][nt * 16 + mn];
    #pragma unroll
    for (int r = 0; r < 8; ++r)
      X[(node * TILE + r + 8 * hi) * 128 + nt * 16 + mn] = (__bf16)(c[r] + bb);
  }
  for (int idx = tid; idx < N_NODES * TILE * 64; idx += 256) {
    int node = idx / (TILE * 64);
    int rem = idx - node * TILE * 64;
    int m = rem >> 6, cc = rem & 63;
    X[(node * TILE + m) * 128 + 64 + cc] =
        (__bf16)p.latent[(((size_t)(b0 + m)) * N_NODES + node) * 64 + cc];
  }
  __syncthreads();

  // ================= Graph convs =================
  conv_layer<8, true >(p, X, Q, V, S, 8, 9, 10, 11);    // gn1: O=128, elu
  conv_layer<4, false>(p, X, Q, V, S, 12, 13, 14, 15);  // gn2: O=64

  // ================= Leg decoder =================
  // L0: gather leg inputs [4][16][256] into Q
  __bf16* LI = Q;
  for (int idx = tid; idx < 4 * TILE * 256; idx += 256) {
    int l = idx >> 12;
    int rem = idx & 4095;
    int m = rem >> 8, cc = rem & 255;
    int part = cc >> 6, c2 = cc & 63;
    int node = (part == 0) ? 0 : (part == 1 ? 1 + l : (part == 2 ? 5 + l : 9 + l));
    LI[idx] = X[(node * TILE + m) * 128 + c2];
  }
  __syncthreads();
  // L1: a1 = elu(LI @ leg_w1 + b1) -> V as [4][16][256]
  __bf16* A1 = V;
  for (int item = wid; item < 4 * 16; item += 8) {
    int leg = item >> 4, nt = item & 15;
    v8f c = zero8();
    const __bf16* A = LI + leg * TILE * 256;
    #pragma unroll
    for (int kt = 0; kt < 8; ++kt)
      c = bwmma(load_a(A, 256, mn, kt, hi), load_w(p.wpk, 16, 16, kt, nt, lane), c);
    float bb = p.bias[16][nt * 16 + mn];
    #pragma unroll
    for (int r = 0; r < 8; ++r)
      A1[(leg * TILE + r + 8 * hi) * 256 + nt * 16 + mn] = (__bf16)eluf(c[r] + bb);
  }
  __syncthreads();
  // L2: a2 = elu(a1 @ leg_w2 + b2) -> Q as [4][16][128]
  __bf16* A2 = Q;
  for (int item = wid; item < 4 * 8; item += 8) {
    int leg = item >> 3, nt = item & 7;
    v8f c = zero8();
    const __bf16* A = A1 + leg * TILE * 256;
    #pragma unroll
    for (int kt = 0; kt < 8; ++kt)
      c = bwmma(load_a(A, 256, mn, kt, hi), load_w(p.wpk, 17, 8, kt, nt, lane), c);
    float bb = p.bias[17][nt * 16 + mn];
    #pragma unroll
    for (int r = 0; r < 8; ++r)
      A2[(leg * TILE + r + 8 * hi) * 128 + nt * 16 + mn] = (__bf16)eluf(c[r] + bb);
  }
  __syncthreads();
  // L3: out = a2 @ leg_w3 + b3 (N padded 3->16; cols>=3 are zero weights)
  for (int item = wid; item < 4; item += 8) {
    int leg = item;
    v8f c = zero8();
    const __bf16* A = A2 + leg * TILE * 128;
    #pragma unroll
    for (int kt = 0; kt < 4; ++kt)
      c = bwmma(load_a(A, 128, mn, kt, hi), load_w(p.wpk, 18, 1, kt, 0, lane), c);
    if (mn < 3) {
      float bb = p.bias[18][mn];
      #pragma unroll
      for (int r = 0; r < 8; ++r)
        p.out[(size_t)(b0 + r + 8 * hi) * 12 + leg * 3 + mn] = c[r] + bb;
    }
  }
}

// Pack f32 [K][N] row-major weight into WMMA B-fragment order (bf16), zero-padded
// to K multiple of 32 and N multiple of 16:
// dst[((kt*NT + nt)*32 + lane)*16 + e] = W[kt*32 + (e>>3)*16 + (lane>>4)*8 + (e&7)][nt*16 + (lane&15)]
__global__ void pack_weights(const float* __restrict__ src, __bf16* __restrict__ dst,
                             int K, int N, int NT, int total) {
  for (int idx = blockIdx.x * blockDim.x + threadIdx.x; idx < total;
       idx += gridDim.x * blockDim.x) {
    int e = idx & 15;
    int lane = (idx >> 4) & 31;
    int t = idx >> 9;
    int nt = t % NT, kt = t / NT;
    int k = kt * 32 + (e >> 3) * 16 + (lane >> 4) * 8 + (e & 7);
    int n = nt * 16 + (lane & 15);
    float v = (k < K && n < N) ? src[k * N + n] : 0.0f;
    dst[idx] = (__bf16)v;
  }
}

extern "C" void kernel_launch(void* const* d_in, const int* in_sizes, int n_in,
                              void* d_out, int out_size, void* d_ws, size_t ws_size,
                              hipStream_t stream) {
  (void)n_in; (void)out_size; (void)ws_size;
  const int B = in_sizes[0] / 111;
  __bf16* wpk = (__bf16*)d_ws;

  static const int widx[19] = {2,4,6,8,10,12,14,16,18,20,22,24,26,28,30,32,34,36,38};
  static const int KK[19]   = {15,128,8,128,8,128,8,128,128,128,128,128,128,128,128,128,256,256,128};
  static const int NN[19]   = {128,64,128,64,128,64,128,64,128,128,128,128,64,64,64,64,256,128,3};
  static const int HOFF[19] = {0,4096,12288,16384,24576,28672,36864,40960,
                               49152,65536,81920,98304,114688,122880,131072,139264,
                               147456,212992,245760};

  for (int i = 0; i < 19; ++i) {
    int NT = (NN[i] + 15) / 16;
    int Kp = (KK[i] + 31) & ~31;
    int total = Kp * NT * 16;
    int blocks = (total + 255) / 256;
    pack_weights<<<blocks, 256, 0, stream>>>((const float*)d_in[widx[i]],
                                             wpk + HOFF[i], KK[i], NN[i], NT, total);
  }

  KParams p;
  p.obs = (const float*)d_in[0];
  p.latent = (const float*)d_in[1];
  p.wpk = wpk;
  for (int i = 0; i < 19; ++i) p.bias[i] = (const float*)d_in[widx[i] + 1];
  p.out = (float*)d_out;

  actor_kernel<<<B / 16, 256, 0, stream>>>(p, B);
}
